// MoeLayer_21603685499663
// MI455X (gfx1250) — compile-verified, matrix operand order checked by reading, not run on hardware
//
#include <hip/hip_runtime.h>

// MoE SwiGLU layer for MI455X (gfx1250, wave32, WMMA + Tensor Data Mover).
// bf16 weights pre-converted/pre-transposed once per launch; expert FFN
// double-buffers TDM tile loads (tensor_load_to_lds) against
// v_wmma_f32_16x16x32_bf16 compute; 32-token M-tiles for L2 efficiency.

typedef __attribute__((ext_vector_type(16))) __bf16 v16bf;
typedef __attribute__((ext_vector_type(8)))  float  v8f;
typedef unsigned int u32x4 __attribute__((ext_vector_type(4)));
typedef int i32x4 __attribute__((ext_vector_type(4)));
typedef int i32x8 __attribute__((ext_vector_type(8)));

#define DM 1024   // d_model
#define FF 4096   // d_ff
#define NE 8      // experts
#define MT 32     // tokens per workgroup tile
// dynamic LDS layout (bytes); ALL shared is dynamic so LDS offset 0 == smem[0]
#define SM_A    0        // X tile: 32 x 1024 bf16                  =  65536
#define SM_H    65536    // h tile: 32 x 128 bf16                   =   8192
#define SM_W1   73728    // W1t tile 128r x 32k bf16, x2 buffers    =  16384
#define SM_W3   90112    // W3t tile 128r x 32k bf16, x2 buffers    =  16384
#define SM_W2   106496   // W2t tile 1024r x 32k bf16, x2 buffers   = 131072
#define SM_TOK  237568   // int sTok[32] + float sWt[32]            =    256
#define SMEM_BYTES 237824

#if defined(__has_builtin)
#if __has_builtin(__builtin_amdgcn_tensor_load_to_lds)
#define HAVE_TDM 1
#endif
#endif
#ifndef HAVE_TDM
#define HAVE_TDM 0
#endif

__device__ __forceinline__ unsigned short f2bf(float f) {
  unsigned int u = __float_as_uint(f);
  u += 0x7FFFu + ((u >> 16) & 1u);           // round-to-nearest-even
  return (unsigned short)(u >> 16);
}

// ---------------- kernel 0: x fp32 -> bf16 ----------------
__global__ void cvt_bf16_kernel(const float* __restrict__ x,
                                unsigned short* __restrict__ xb, int n) {
  int i = blockIdx.x * blockDim.x + threadIdx.x;
  int stride = gridDim.x * blockDim.x;
  for (; i < n; i += stride) xb[i] = f2bf(x[i]);
}

// ------------- kernel 0b: per-expert fp32 [R][C] -> bf16 [C][R] -------------
__global__ void transpose_cvt_kernel(const float* __restrict__ src,
                                     unsigned short* __restrict__ dst,
                                     int R, int C) {
  __shared__ float tile[32][33];
  const int e = blockIdx.z;
  const float* s = src + (size_t)e * R * C;
  unsigned short* d = dst + (size_t)e * R * C;
  const int c0 = blockIdx.x * 32, r0 = blockIdx.y * 32;
  const int tx = threadIdx.x, ty = threadIdx.y;          // 32 x 8
#pragma unroll
  for (int i = 0; i < 32; i += 8)
    tile[ty + i][tx] = s[(size_t)(r0 + ty + i) * C + (c0 + tx)];
  __syncthreads();
#pragma unroll
  for (int i = 0; i < 32; i += 8)
    d[(size_t)(c0 + ty + i) * R + (r0 + tx)] = f2bf(tile[tx][ty + i]);
}

// ---------------- kernel 1: gating, one wave32 per token ----------------
__global__ void gate_kernel(const float* __restrict__ x, const float* __restrict__ gw,
                            int* __restrict__ cnt, int* __restrict__ tokl,
                            float* __restrict__ wtl, int T) {
  int wave = threadIdx.x >> 5;
  int lane = threadIdx.x & 31;
  int t = blockIdx.x * (blockDim.x >> 5) + wave;
  if (t >= T) return;
  const float* xr = x + (size_t)t * DM;
  float acc[NE];
#pragma unroll
  for (int e = 0; e < NE; ++e) acc[e] = 0.f;
  for (int d = lane; d < DM; d += 32) {
    float xv = xr[d];
    const float* g = gw + (size_t)d * NE;
#pragma unroll
    for (int e = 0; e < NE; ++e) acc[e] += xv * g[e];
  }
#pragma unroll
  for (int off = 16; off > 0; off >>= 1) {
#pragma unroll
    for (int e = 0; e < NE; ++e) acc[e] += __shfl_xor(acc[e], off);
  }
  if (lane == 0) {
    int e0 = 0; float l0 = acc[0];
#pragma unroll
    for (int e = 1; e < NE; ++e) if (acc[e] > l0) { l0 = acc[e]; e0 = e; }
    int e1 = (e0 == 0) ? 1 : 0; float l1 = acc[e1];
#pragma unroll
    for (int e = 0; e < NE; ++e) if (e != e0 && acc[e] > l1) { l1 = acc[e]; e1 = e; }
    float p1 = __expf(l1 - l0);                       // l0 >= l1
    float inv = __builtin_amdgcn_rcpf(1.f + p1);
    float w0 = inv, w1 = p1 * inv;
    int p = atomicAdd(&cnt[e0], 1);
    tokl[(size_t)e0 * T + p] = t; wtl[(size_t)e0 * T + p] = w0;
    p = atomicAdd(&cnt[e1], 1);
    tokl[(size_t)e1 * T + p] = t; wtl[(size_t)e1 * T + p] = w1;
  }
}

// ---- TDM: stage a 2D bf16 tile (rows x 32 elems, row stride strideElems) ----
// LDS result is contiguous [row][32 elems] at byte offset ldsoff.
__device__ __forceinline__ void tdm_issue_2d(unsigned ldsoff, const void* gptr,
                                             unsigned tileRows, unsigned strideElems) {
#if HAVE_TDM
  unsigned long long ga = (unsigned long long)gptr;
  u32x4 g0;
  g0[0] = 1u;                                              // count=1, user mode
  g0[1] = ldsoff;                                          // lds_addr (bytes)
  g0[2] = (unsigned)ga;                                    // global_addr[31:0]
  g0[3] = (unsigned)((ga >> 32) & 0x01FFFFFFu) | 0x80000000u; // ga[56:32] | type=2
  i32x8 g1;
  g1[0] = 0x00010000;                  // wg_mask=0, data_size=2B, no flags
  g1[1] = (int)0xFFFF0000u;            // tensor_dim0 = 0x7FFFFFFF (low16<<16)
  g1[2] = (int)(0x7FFFu | 0xFFFF0000u);// dim0 hi | tensor_dim1=0x7FFFFFFF lo<<16
  g1[3] = (int)(0x7FFFu | (32u << 16));// dim1 hi | tile_dim0 = 32 elems
  g1[4] = (int)(tileRows & 0xFFFFu);   // tile_dim1 = rows, tile_dim2 = 0
  g1[5] = (int)strideElems;            // tensor_dim0_stride[31:0]
  g1[6] = 0;
  g1[7] = 0;
  i32x4 z4 = {0, 0, 0, 0};
  i32x8 z8 = {0, 0, 0, 0, 0, 0, 0, 0};
  __builtin_amdgcn_tensor_load_to_lds(g0, g1, z4, z4, z8, 0);
#else
  (void)ldsoff; (void)gptr; (void)tileRows; (void)strideElems;
#endif
}

// fallback cooperative copy (bf16 -> LDS, no conversion needed)
__device__ __forceinline__ void coop_stage_2d(char* smem, unsigned ldsoff,
                                              const unsigned short* gptr,
                                              int rows, int strideElems) {
  int nvec = rows * 4;                          // 16B vectors (64B per row)
  uint4* dst = (uint4*)(smem + ldsoff);
  for (int i = threadIdx.x; i < nvec; i += 256) {
    int row = i >> 2, seg = i & 3;
    dst[i] = *(const uint4*)(gptr + (size_t)row * strideElems + seg * 8);
  }
}

// ---------------- kernel 2: fused expert FFN (SwiGLU) ----------------
// 256 threads = 8 waves; one 32-token tile of one expert per block.
// Wave w owns out columns [w*128, w*128+128): 2x8 persistent 16x16 f32 accs.
__global__ void __launch_bounds__(256)
moe_expert_kernel(const unsigned short* __restrict__ xb,
                  const unsigned short* __restrict__ w1t,   // [E][F][D] bf16
                  const unsigned short* __restrict__ w3t,   // [E][F][D] bf16
                  const unsigned short* __restrict__ w2t,   // [E][D][F] bf16
                  const int* __restrict__ cnt, const int* __restrict__ tokl,
                  const float* __restrict__ wtl, float* __restrict__ out, int T) {
  extern __shared__ __align__(16) char smem[];
  int*   sTok = (int*)(smem + SM_TOK);
  float* sWt  = (float*)(smem + SM_TOK + 128);

  const int tilesPerE = T / MT;
  const int e    = blockIdx.x / tilesPerE;
  const int tile = blockIdx.x % tilesPerE;
  const int ne   = cnt[e];
  if (tile * MT >= ne) return;               // block-uniform exit

  if (threadIdx.x < MT) {
    int r = tile * MT + threadIdx.x;
    if (r < ne) { sTok[threadIdx.x] = tokl[(size_t)e * T + r];
                  sWt[threadIdx.x]  = wtl[(size_t)e * T + r]; }
    else        { sTok[threadIdx.x] = 0; sWt[threadIdx.x] = 0.f; }  // weight-0 pad
  }
  __syncthreads();

  // gather 32 token rows (bf16) into LDS Abuf[32][1024]
  {
    int row = threadIdx.x >> 3, seg = threadIdx.x & 7;   // 8 threads per row
    const uint4* src = (const uint4*)(xb + (size_t)sTok[row] * DM);
    uint4* dst = (uint4*)(smem + SM_A + row * (DM * 2));
#pragma unroll
    for (int i = 0; i < 16; ++i) dst[seg * 16 + i] = src[seg * 16 + i];
  }
  __syncthreads();

  const int wv   = threadIdx.x >> 5;
  const int lane = threadIdx.x & 31;
  const int nn   = lane & 15;
  const int hb   = lane >> 4;

  const unsigned short* W1e = w1t + (size_t)e * FF * DM;  // [F][D]
  const unsigned short* W3e = w3t + (size_t)e * FF * DM;  // [F][D]
  const unsigned short* W2e = w2t + (size_t)e * DM * FF;  // [D][F]

  unsigned short* hbf = (unsigned short*)(smem + SM_H);
  const char* abase = smem + SM_A;

  v8f accO[2][8];
#pragma unroll
  for (int m = 0; m < 2; ++m)
#pragma unroll
    for (int j = 0; j < 8; ++j)
      accO[m][j] = (v8f){0.f,0.f,0.f,0.f,0.f,0.f,0.f,0.f};

  for (int fc = 0; fc < FF; fc += 128) {       // 32 F-chunks
    v8f acc1[2], acc3[2];
#pragma unroll
    for (int m = 0; m < 2; ++m) {
      acc1[m] = (v8f){0.f,0.f,0.f,0.f,0.f,0.f,0.f,0.f};
      acc3[m] = (v8f){0.f,0.f,0.f,0.f,0.f,0.f,0.f,0.f};
    }

    // ---- stage 1: X@W1 and X@W3 over per-wave 16-column slice, double-buffered
#if HAVE_TDM
    if (wv == 0) {
      tdm_issue_2d(SM_W1, W1e + (size_t)fc * DM, 128, DM);
      tdm_issue_2d(SM_W3, W3e + (size_t)fc * DM, 128, DM);
    }
#endif
    for (int i = 0; i < 32; ++i) {
      const int k0 = i * 32;
#if HAVE_TDM
      if (wv == 0) {
        if (i < 31) {
          unsigned b = (unsigned)((i + 1) & 1) * 8192u;
          tdm_issue_2d(SM_W1 + b, W1e + (size_t)fc * DM + (k0 + 32), 128, DM);
          tdm_issue_2d(SM_W3 + b, W3e + (size_t)fc * DM + (k0 + 32), 128, DM);
          __builtin_amdgcn_s_wait_tensorcnt(2);  // tile i complete; i+1 in flight
        } else {
          __builtin_amdgcn_s_wait_tensorcnt(0);
        }
      }
#else
      {
        unsigned b = (unsigned)(i & 1) * 8192u;
        coop_stage_2d(smem, SM_W1 + b, W1e + (size_t)fc * DM + k0, 128, DM);
        coop_stage_2d(smem, SM_W3 + b, W3e + (size_t)fc * DM + k0, 128, DM);
      }
#endif
      __syncthreads();
      const unsigned cb = (unsigned)(i & 1) * 8192u;
      union { uint4 q[2]; v16bf v; } A0, A1, B1, B3;
      const char* ar0 = abase + (0 * 16 + nn) * (DM * 2) + (size_t)(k0 + hb * 8) * 2;
      const char* ar1 = abase + (1 * 16 + nn) * (DM * 2) + (size_t)(k0 + hb * 8) * 2;
      A0.q[0] = *(const uint4*)(ar0);  A0.q[1] = *(const uint4*)(ar0 + 32);
      A1.q[0] = *(const uint4*)(ar1);  A1.q[1] = *(const uint4*)(ar1 + 32);
      const char* b1p = smem + SM_W1 + cb + (size_t)(wv * 16 + nn) * 64 + hb * 16;
      B1.q[0] = *(const uint4*)(b1p);  B1.q[1] = *(const uint4*)(b1p + 32);
      const char* b3p = smem + SM_W3 + cb + (size_t)(wv * 16 + nn) * 64 + hb * 16;
      B3.q[0] = *(const uint4*)(b3p);  B3.q[1] = *(const uint4*)(b3p + 32);
      acc1[0] = __builtin_amdgcn_wmma_f32_16x16x32_bf16(false, A0.v, false, B1.v,
                                                        (short)0, acc1[0], false, false);
      acc1[1] = __builtin_amdgcn_wmma_f32_16x16x32_bf16(false, A1.v, false, B1.v,
                                                        (short)0, acc1[1], false, false);
      acc3[0] = __builtin_amdgcn_wmma_f32_16x16x32_bf16(false, A0.v, false, B3.v,
                                                        (short)0, acc3[0], false, false);
      acc3[1] = __builtin_amdgcn_wmma_f32_16x16x32_bf16(false, A1.v, false, B3.v,
                                                        (short)0, acc3[1], false, false);
      __syncthreads();
    }

    // ---- SwiGLU in registers, h tile (32x128 bf16) -> LDS
#pragma unroll
    for (int m = 0; m < 2; ++m) {
#pragma unroll
      for (int v = 0; v < 8; ++v) {
        int M = m * 16 + v + hb * 8;
        float g = acc1[m][v], u = acc3[m][v];
        float sig = __builtin_amdgcn_rcpf(1.f + __expf(-g));
        hbf[M * 128 + wv * 16 + nn] = f2bf(g * sig * u);
      }
    }
#if HAVE_TDM
    if (wv == 0) tdm_issue_2d(SM_W2, W2e + (size_t)fc, 1024, FF);
#endif
    __syncthreads();

    // ---- stage 2: accO += h_chunk @ W2[fc:fc+128, :], double-buffered
    for (int i = 0; i < 4; ++i) {
      const int kk0 = i * 32;
#if HAVE_TDM
      if (wv == 0) {
        if (i < 3) {
          unsigned b = (unsigned)((i + 1) & 1) * 65536u;
          tdm_issue_2d(SM_W2 + b, W2e + (size_t)(fc + kk0 + 32), 1024, FF);
          __builtin_amdgcn_s_wait_tensorcnt(1);
        } else {
          __builtin_amdgcn_s_wait_tensorcnt(0);
        }
      }
#else
      {
        unsigned b = (unsigned)(i & 1) * 65536u;
        coop_stage_2d(smem, SM_W2 + b, W2e + (size_t)(fc + kk0), 1024, FF);
      }
#endif
      __syncthreads();
      const unsigned cb = (unsigned)(i & 1) * 65536u;
      union { uint4 q[2]; v16bf v; } Ah0, Ah1;
      const char* hr0 = (const char*)hbf + (0 * 16 + nn) * 256 + (size_t)(kk0 + hb * 8) * 2;
      const char* hr1 = (const char*)hbf + (1 * 16 + nn) * 256 + (size_t)(kk0 + hb * 8) * 2;
      Ah0.q[0] = *(const uint4*)(hr0);  Ah0.q[1] = *(const uint4*)(hr0 + 32);
      Ah1.q[0] = *(const uint4*)(hr1);  Ah1.q[1] = *(const uint4*)(hr1 + 32);
#pragma unroll
      for (int j = 0; j < 8; ++j) {
        union { uint4 q[2]; v16bf v; } B2;
        const char* bp = smem + SM_W2 + cb +
                         (size_t)(wv * 128 + j * 16 + nn) * 64 + hb * 16;
        B2.q[0] = *(const uint4*)(bp);  B2.q[1] = *(const uint4*)(bp + 32);
        accO[0][j] = __builtin_amdgcn_wmma_f32_16x16x32_bf16(false, Ah0.v, false, B2.v,
                                                             (short)0, accO[0][j], false, false);
        accO[1][j] = __builtin_amdgcn_wmma_f32_16x16x32_bf16(false, Ah1.v, false, B2.v,
                                                             (short)0, accO[1][j], false, false);
      }
      __syncthreads();
    }
  }

  // weighted scatter: exactly 2 f32 atomic adds per output element -> deterministic
#pragma unroll
  for (int m = 0; m < 2; ++m) {
#pragma unroll
    for (int j = 0; j < 8; ++j) {
      int d = wv * 128 + j * 16 + nn;
#pragma unroll
      for (int v = 0; v < 8; ++v) {
        int M = m * 16 + v + hb * 8;
        atomicAdd(out + (size_t)sTok[M] * DM + d, sWt[M] * accO[m][j][v]);
      }
    }
  }
}

extern "C" void kernel_launch(void* const* d_in, const int* in_sizes, int n_in,
                              void* d_out, int out_size, void* d_ws, size_t ws_size,
                              hipStream_t stream) {
  const float* x  = (const float*)d_in[0];
  const float* gw = (const float*)d_in[1];
  const float* w1 = (const float*)d_in[2];
  const float* w2 = (const float*)d_in[3];
  const float* w3 = (const float*)d_in[4];
  float* out = (float*)d_out;
  const int T = in_sizes[0] / DM;            // 4096 tokens
  const size_t WSZ = (size_t)NE * DM * FF * 2;   // bf16 weight copy, per tensor

  // workspace layout
  size_t off = 0;
  unsigned short* xb  = (unsigned short*)((char*)d_ws + off);
  off += (size_t)T * DM * 2;            off = (off + 255) & ~(size_t)255;
  unsigned short* w1t = (unsigned short*)((char*)d_ws + off); off += WSZ;
  unsigned short* w3t = (unsigned short*)((char*)d_ws + off); off += WSZ;
  unsigned short* w2t = (unsigned short*)((char*)d_ws + off); off += WSZ;
  int* cntp = (int*)((char*)d_ws + off);
  off += NE * sizeof(int);              off = (off + 255) & ~(size_t)255;
  int* tokl = (int*)((char*)d_ws + off);
  off += (size_t)NE * T * sizeof(int);  off = (off + 255) & ~(size_t)255;
  float* wtl = (float*)((char*)d_ws + off);

  (void)hipMemsetAsync(cntp, 0, NE * sizeof(int), stream);
  (void)hipMemsetAsync(d_out, 0, (size_t)out_size * sizeof(float), stream);

  cvt_bf16_kernel<<<1024, 256, 0, stream>>>(x, xb, T * DM);
  {
    dim3 blk(32, 8);
    dim3 g13(FF / 32, DM / 32, NE);   // w1/w3: [D][F] -> [F][D]
    dim3 g2 (DM / 32, FF / 32, NE);   // w2:    [F][D] -> [D][F]
    transpose_cvt_kernel<<<g13, blk, 0, stream>>>(w1, w1t, DM, FF);
    transpose_cvt_kernel<<<g13, blk, 0, stream>>>(w3, w3t, DM, FF);
    transpose_cvt_kernel<<<g2,  blk, 0, stream>>>(w2, w2t, FF, DM);
  }
  gate_kernel<<<(T + 7) / 8, 256, 0, stream>>>(x, gw, cntp, tokl, wtl, T);

  (void)hipFuncSetAttribute((const void*)moe_expert_kernel,
                            hipFuncAttributeMaxDynamicSharedMemorySize, SMEM_BYTES);
  const int tilesPerE = T / MT;
  moe_expert_kernel<<<NE * tilesPerE, 256, SMEM_BYTES, stream>>>(
      xb, w1t, w3t, w2t, cntp, tokl, wtl, out, T);
}
